// MLP_Route_RL_Model_42073499631626
// MI455X (gfx1250) — matrix operand order repaired and checked by reading, not compiled, not run on hardware
//
#include <hip/hip_runtime.h>
#include <hip/hip_bf16.h>

// ---------------- types / helpers ----------------
typedef __attribute__((ext_vector_type(16))) _Float16 v16h;
typedef __attribute__((ext_vector_type(8)))  _Float16 v8h;
typedef __attribute__((ext_vector_type(8)))  float    v8f;
typedef int av4i __attribute__((vector_size(16))); // matches async-builtin param

#define LANE (threadIdx.x & 31)
#define WAVE (threadIdx.x >> 5)

constexpr int Bb   = 256;
constexpr int MAXR = 48;
constexpr int MAXN = 24;
constexpr int FEAT = 36;
constexpr int NPAIR = Bb * MAXR;          // 12288
constexpr int NROW  = NPAIR * MAXN;       // 294912
constexpr int SSTRIDE = MAXR + MAXR * MAXN * FEAT; // 41520 floats per batch row

// CDNA5 async global->LDS path (guarded: fall back to b128 copy loop if the
// toolchain lacks the builtins; host pass always takes the fallback branch).
#if defined(__gfx1250__) && __has_builtin(__builtin_amdgcn_global_load_async_to_lds_b128) && __has_builtin(__builtin_amdgcn_s_wait_asynccnt)
#define USE_ASYNC_LDS 1
#else
#define USE_ASYNC_LDS 0
#endif

// ---- workspace layout (bytes) ----
constexpr size_t WS_WC1  = 0;                      // 64x128  -> 16 frags (16KB)
constexpr size_t WS_WC2  = WS_WC1  + 16 * 1024;    // 128x32  ->  8 frags (8KB)
constexpr size_t WS_WIH0 = WS_WC2  + 8 * 1024;     // 32x384  -> 24 frags (24KB)
constexpr size_t WS_WHH0 = WS_WIH0 + 24 * 1024;    // 128x384 -> 96 frags (96KB)
constexpr size_t WS_WIH1 = WS_WHH0 + 96 * 1024;    // 128x384 (96KB)
constexpr size_t WS_WHH1 = WS_WIH1 + 96 * 1024;    // 128x384 (96KB)
constexpr size_t WS_WN1  = WS_WHH1 + 96 * 1024;    // 256x256 -> 128 frags (128KB)
constexpr size_t WS_WN2  = WS_WN1  + 128 * 1024;   // 256x128 -> 64 frags (64KB)
constexpr size_t WS_WN3  = WS_WN2  + 64 * 1024;    // 128x32  ->  8 frags (8KB)
constexpr size_t WS_XSEQ = 1u << 20;                               // f16 [24][12288][32]
constexpr size_t WS_H2   = WS_XSEQ + (size_t)24 * NPAIR * 32 * 2;  // f32 [12288][128]
constexpr size_t WS_MEAN = WS_H2   + (size_t)NPAIR * 128 * 4;      // f32 [256][128]

constexpr int GRU_WBYTES = (24 + 96 + 96 + 96) * 512 * 2;          // 319488

__device__ __forceinline__ v8f wmma16(v16h a, v16h b, v8f c) {
  return __builtin_amdgcn_wmma_f32_16x16x32_f16(false, a, false, b, (short)0, c,
                                                false, false);
}

__device__ __forceinline__ v8f splat8(float x) {
  v8f r;
#pragma unroll
  for (int e = 0; e < 8; ++e) r[e] = x;
  return r;
}

// Branch-free activations: native v_exp_f32 / v_rcp_f32, no EXEC manipulation.
__device__ __forceinline__ float sigmf(float x) {
  return __builtin_amdgcn_rcpf(1.0f + __expf(-x));
}
__device__ __forceinline__ float tanhfast(float x) {
  // tanh(x) = 1 - 2/(exp(2x)+1); exp overflow/underflow saturates correctly.
  return 1.0f - 2.0f * __builtin_amdgcn_rcpf(__expf(2.0f * x) + 1.0f);
}

// B fragment: lane l holds col (l&15), K = (l>=16?16:0)+e, e = 0..15.
// Swizzled storage: frag*512 + lane*16 + e  (halves) -> two coalesced b128 loads.
__device__ __forceinline__ v16h load_bfrag(const _Float16* frag) {
  const _Float16* p = frag + (size_t)LANE * 16;
  v8h lo = *(const v8h*)(p);
  v8h hi = *(const v8h*)(p + 8);
  return __builtin_shufflevector(lo, hi, 0, 1, 2, 3, 4, 5, 6, 7,
                                 8, 9, 10, 11, 12, 13, 14, 15);
}

// A fragment (16x32 f16, row-major source, row stride = strideH halves):
// lane l: row = l&15, kb = k0 + (l>>4)*16; halves kb..kb+15.
__device__ __forceinline__ v16h load_afrag(const _Float16* base, int strideH, int k0) {
  int l = LANE;
  const _Float16* p = base + (size_t)(l & 15) * strideH + k0 + ((l >> 4) << 4);
  v8h lo = *(const v8h*)(p);
  v8h hi = *(const v8h*)(p + 8);
  return __builtin_shufflevector(lo, hi, 0, 1, 2, 3, 4, 5, 6, 7,
                                 8, 9, 10, 11, 12, 13, 14, 15);
}

// ---------------- weight swizzle (f32 row-major -> f16 fragment-major) ------
__global__ __launch_bounds__(256) void swizzle_w(const float* __restrict__ W,
                                                 _Float16* __restrict__ out,
                                                 int K, int Kpad, int N, int Npad) {
  int idx = blockIdx.x * blockDim.x + threadIdx.x;
  if (idx >= Kpad * Npad) return;
  int k = idx / Npad, n = idx % Npad;
  float v = (k < K && n < N) ? W[(size_t)k * N + n] : 0.0f;
  int kc = k >> 5, kh = (k >> 4) & 1, e = k & 15;
  int nt = n >> 4, lane = (kh << 4) | (n & 15);
  int frag = nt * (Kpad >> 5) + kc;
  out[(size_t)frag * 512 + lane * 16 + e] = (_Float16)v;
}

// ---------------- kernel A: customer MLP -----------------------------------
__global__ __launch_bounds__(256) void cust_mlp_kernel(
    const float* __restrict__ state, const _Float16* __restrict__ wc1s,
    const float* __restrict__ bc1, const _Float16* __restrict__ wc2s,
    const float* __restrict__ bc2, _Float16* __restrict__ xseq) {
  extern __shared__ _Float16 smemA[];
  _Float16* sA = smemA + WAVE * (16 * 64);                  // 16x64 padded input
  _Float16* sH = smemA + 8 * (16 * 64) + WAVE * (16 * 128); // 16x128 hidden
  const int l = LANE, col15 = l & 15, rb = (l >> 4) * 8;
  const int rowbase = (blockIdx.x * 8 + WAVE) * 16;

  for (int i = l; i < 16 * 64; i += 32) {
    int r = i >> 6, k = i & 63;
    int rg = rowbase + r;
    float v = 0.0f;
    if (k < FEAT) {
      int b = rg / (MAXR * MAXN);
      int rem = rg % (MAXR * MAXN); // route*24 + node
      v = state[(size_t)b * SSTRIDE + MAXR + (size_t)rem * FEAT + k];
    }
    sA[i] = (_Float16)v;
  }

#pragma unroll
  for (int j = 0; j < 8; ++j) { // h = tanh(A @ Wc1 + bc1)
    v8f acc = splat8(bc1[j * 16 + col15]);
#pragma unroll
    for (int kc = 0; kc < 2; ++kc) {
      v16h a = load_afrag(sA, 64, kc * 32);
      v16h b = load_bfrag(wc1s + (size_t)(j * 2 + kc) * 512);
      acc = wmma16(a, b, acc);
    }
    int col = j * 16 + col15;
#pragma unroll
    for (int e = 0; e < 8; ++e) sH[(rb + e) * 128 + col] = (_Float16)tanhfast(acc[e]);
  }

#pragma unroll
  for (int j = 0; j < 2; ++j) { // x = tanh(h @ Wc2 + bc2) -> [node][pair][32] f16
    v8f acc = splat8(bc2[j * 16 + col15]);
#pragma unroll
    for (int kc = 0; kc < 4; ++kc) {
      v16h a = load_afrag(sH, 128, kc * 32);
      v16h b = load_bfrag(wc2s + (size_t)(j * 4 + kc) * 512);
      acc = wmma16(a, b, acc);
    }
    int col = j * 16 + col15;
#pragma unroll
    for (int e = 0; e < 8; ++e) {
      int rg = rowbase + rb + e;
      int node = rg % MAXN, pair = rg / MAXN;
      xseq[((size_t)node * NPAIR + pair) * 32 + col] = (_Float16)tanhfast(acc[e]);
    }
  }
}

// ---------------- kernel B: 2-layer GRU, 24 steps --------------------------
__device__ __forceinline__ void stage_chunk(_Float16* stg, const v8f* h, int kc) {
  int l = LANE, col15 = l & 15, rb = (l >> 4) * 8;
#pragma unroll
  for (int p = 0; p < 2; ++p) {
    const v8f t = h[2 * kc + p];
#pragma unroll
    for (int e = 0; e < 8; ++e)
      stg[(rb + e) * 32 + p * 16 + col15] = (_Float16)t[e];
  }
}

__global__ __launch_bounds__(128) void gru_kernel(
    const _Float16* __restrict__ xseq, const _Float16* __restrict__ wih0s,
    const float* __restrict__ bih0, const float* __restrict__ bhh0,
    const float* __restrict__ bih1, const float* __restrict__ bhh1,
    float* __restrict__ h2out) {
  extern __shared__ _Float16 smemB[];
  // 312KB of swizzled weights contiguous in ws: Wih0(24) Whh0(96) Wih1(96) Whh1(96)
  _Float16* Lih0 = smemB;
  _Float16* Lhh0 = Lih0 + 24 * 512;
  _Float16* Lih1 = Lhh0 + 96 * 512;
  _Float16* Lhh1 = Lih1 + 96 * 512;
  _Float16* Lstg = Lhh1 + 96 * 512 + WAVE * (16 * 32); // 1KB per wave

#if USE_ASYNC_LDS
  { // ASYNCcnt-tracked DMA into LDS, bypasses VGPRs (b128 per lane)
    const char* s = (const char*)wih0s;
    char* d = (char*)smemB;
    for (int i = (int)threadIdx.x * 16; i < GRU_WBYTES; i += (int)blockDim.x * 16) {
      __builtin_amdgcn_global_load_async_to_lds_b128(
          (__attribute__((address_space(1))) av4i*)(s + i),
          (__attribute__((address_space(3))) av4i*)(d + i), 0, 0);
    }
    __builtin_amdgcn_s_wait_asynccnt(0);
  }
#else
  {
    const uint4* s = (const uint4*)wih0s;
    uint4* d = (uint4*)smemB;
    for (int i = threadIdx.x; i < GRU_WBYTES / 16; i += blockDim.x) d[i] = s[i];
  }
#endif
  __syncthreads();

  const int l = LANE, col15 = l & 15, rb = (l >> 4) * 8;
  const int pairbase = blockIdx.x * 64 + WAVE * 16;

  v8f h1f[8], h2f[8];
#pragma unroll
  for (int j = 0; j < 8; ++j) { h1f[j] = splat8(0.0f); h2f[j] = splat8(0.0f); }

  for (int t = 0; t < MAXN; ++t) {
    const _Float16* xA = xseq + ((size_t)t * NPAIR + pairbase) * 32;
    if (t + 1 < MAXN)
      __builtin_prefetch(xseq + ((size_t)(t + 1) * NPAIR + pairbase) * 32, 0, 3);

    // -------- layer 0: h1 = GRU(x_t, h1) --------
    v8f arz[16], ain[8], ahn[8];
#pragma unroll
    for (int j = 0; j < 16; ++j)
      arz[j] = splat8(bih0[j * 16 + col15] + bhh0[j * 16 + col15]);
#pragma unroll
    for (int j = 0; j < 8; ++j) {
      ain[j] = splat8(bih0[256 + j * 16 + col15]);
      ahn[j] = splat8(bhh0[256 + j * 16 + col15]);
    }
    { // gi = x @ Wih0 (K=32, A straight from global time-major x)
      v16h a = load_afrag(xA, 32, 0);
#pragma unroll
      for (int j = 0; j < 16; ++j)
        arz[j] = wmma16(a, load_bfrag(Lih0 + (size_t)j * 512), arz[j]);
#pragma unroll
      for (int j = 0; j < 8; ++j)
        ain[j] = wmma16(a, load_bfrag(Lih0 + (size_t)(16 + j) * 512), ain[j]);
    }
#pragma unroll
    for (int kc = 0; kc < 4; ++kc) { // gh = h1 @ Whh0
      stage_chunk(Lstg, h1f, kc);
      v16h a = load_afrag(Lstg, 32, 0);
#pragma unroll
      for (int j = 0; j < 16; ++j)
        arz[j] = wmma16(a, load_bfrag(Lhh0 + (size_t)(j * 4 + kc) * 512), arz[j]);
#pragma unroll
      for (int j = 0; j < 8; ++j)
        ahn[j] = wmma16(a, load_bfrag(Lhh0 + (size_t)((16 + j) * 4 + kc) * 512), ahn[j]);
    }
#pragma unroll
    for (int j = 0; j < 8; ++j)
#pragma unroll
      for (int e = 0; e < 8; ++e) {
        float r = sigmf(arz[j][e]);
        float z = sigmf(arz[8 + j][e]);
        float n = tanhfast(ain[j][e] + r * ahn[j][e]);
        h1f[j][e] = (1.0f - z) * n + z * h1f[j][e];
      }

    // -------- layer 1: h2 = GRU(h1, h2) --------
#pragma unroll
    for (int j = 0; j < 16; ++j)
      arz[j] = splat8(bih1[j * 16 + col15] + bhh1[j * 16 + col15]);
#pragma unroll
    for (int j = 0; j < 8; ++j) {
      ain[j] = splat8(bih1[256 + j * 16 + col15]);
      ahn[j] = splat8(bhh1[256 + j * 16 + col15]);
    }
#pragma unroll
    for (int kc = 0; kc < 4; ++kc) { // gi = h1 @ Wih1
      stage_chunk(Lstg, h1f, kc);
      v16h a = load_afrag(Lstg, 32, 0);
#pragma unroll
      for (int j = 0; j < 16; ++j)
        arz[j] = wmma16(a, load_bfrag(Lih1 + (size_t)(j * 4 + kc) * 512), arz[j]);
#pragma unroll
      for (int j = 0; j < 8; ++j)
        ain[j] = wmma16(a, load_bfrag(Lih1 + (size_t)((16 + j) * 4 + kc) * 512), ain[j]);
    }
#pragma unroll
    for (int kc = 0; kc < 4; ++kc) { // gh = h2 @ Whh1
      stage_chunk(Lstg, h2f, kc);
      v16h a = load_afrag(Lstg, 32, 0);
#pragma unroll
      for (int j = 0; j < 16; ++j)
        arz[j] = wmma16(a, load_bfrag(Lhh1 + (size_t)(j * 4 + kc) * 512), arz[j]);
#pragma unroll
      for (int j = 0; j < 8; ++j)
        ahn[j] = wmma16(a, load_bfrag(Lhh1 + (size_t)((16 + j) * 4 + kc) * 512), ahn[j]);
    }
#pragma unroll
    for (int j = 0; j < 8; ++j)
#pragma unroll
      for (int e = 0; e < 8; ++e) {
        float r = sigmf(arz[j][e]);
        float z = sigmf(arz[8 + j][e]);
        float n = tanhfast(ain[j][e] + r * ahn[j][e]);
        h2f[j][e] = (1.0f - z) * n + z * h2f[j][e];
      }
  }

#pragma unroll
  for (int j = 0; j < 8; ++j) {
    int col = j * 16 + col15;
#pragma unroll
    for (int e = 0; e < 8; ++e)
      h2out[(size_t)(pairbase + rb + e) * 128 + col] = h2f[j][e];
  }
}

// ---------------- mean over routes -----------------------------------------
__global__ __launch_bounds__(128) void mean_kernel(const float* __restrict__ h2,
                                                   float* __restrict__ mean) {
  int b = blockIdx.x, c = threadIdx.x;
  float s = 0.0f;
  for (int r = 0; r < MAXR; ++r) s += h2[(size_t)(b * MAXR + r) * 128 + c];
  mean[(size_t)b * 128 + c] = s * (1.0f / (float)MAXR);
}

// ---------------- kernel C: node head + masked softmax ----------------------
__global__ __launch_bounds__(128) void head_kernel(
    const float* __restrict__ h2, const float* __restrict__ mean,
    const float* __restrict__ state, const _Float16* __restrict__ wn1s,
    const float* __restrict__ bn1, const _Float16* __restrict__ wn2s,
    const float* __restrict__ bn2, const _Float16* __restrict__ wn3s,
    const float* __restrict__ bn3, float* __restrict__ out) {
  extern __shared__ char smemC[];
  _Float16* sAall = (_Float16*)smemC;        // 4 * 16x256
  _Float16* sHall = sAall + 4 * 4096;        // 4 * 16x256
  float* sLall = (float*)(sHall + 4 * 4096); // 4 * 16x32 logits
  _Float16* sA = sAall + WAVE * 4096;
  _Float16* sH = sHall + WAVE * 4096;
  float* sL = sLall + WAVE * 512;

  const int l = LANE, col15 = l & 15, rb = (l >> 4) * 8;
  const int pairbase = blockIdx.x * 64 + WAVE * 16;

  for (int i = l; i < 16 * 256; i += 32) { // node_in = [h2 | mean] as 16x256 f16
    int r = i >> 8, c = i & 255;
    int pair = pairbase + r;
    float v = (c < 128) ? h2[(size_t)pair * 128 + c]
                        : mean[(size_t)(pair / MAXR) * 128 + (c - 128)];
    sA[i] = (_Float16)v;
  }

#pragma unroll
  for (int j = 0; j < 16; ++j) { // relu(node_in @ Wn1 + bn1) (256 -> 256)
    v8f acc = splat8(bn1[j * 16 + col15]);
#pragma unroll
    for (int kc = 0; kc < 8; ++kc) {
      v16h a = load_afrag(sA, 256, kc * 32);
      v16h b = load_bfrag(wn1s + (size_t)(j * 8 + kc) * 512);
      acc = wmma16(a, b, acc);
    }
    int col = j * 16 + col15;
#pragma unroll
    for (int e = 0; e < 8; ++e)
      sH[(rb + e) * 256 + col] = (_Float16)fmaxf(acc[e], 0.0f);
  }

#pragma unroll
  for (int j = 0; j < 8; ++j) { // relu(h @ Wn2 + bn2) (256 -> 128) -> sA cols 0..127
    v8f acc = splat8(bn2[j * 16 + col15]);
#pragma unroll
    for (int kc = 0; kc < 8; ++kc) {
      v16h a = load_afrag(sH, 256, kc * 32);
      v16h b = load_bfrag(wn2s + (size_t)(j * 8 + kc) * 512);
      acc = wmma16(a, b, acc);
    }
    int col = j * 16 + col15;
#pragma unroll
    for (int e = 0; e < 8; ++e)
      sA[(rb + e) * 256 + col] = (_Float16)fmaxf(acc[e], 0.0f);
  }

#pragma unroll
  for (int j = 0; j < 2; ++j) { // logits = h @ Wn3 + bn3 (128 -> 24, N pad 32)
    int cg = j * 16 + col15;
    v8f acc = splat8(cg < MAXN ? bn3[cg] : 0.0f);
#pragma unroll
    for (int kc = 0; kc < 4; ++kc) {
      v16h a = load_afrag(sA, 256, kc * 32);
      v16h b = load_bfrag(wn3s + (size_t)(j * 4 + kc) * 512);
      acc = wmma16(a, b, acc);
    }
#pragma unroll
    for (int e = 0; e < 8; ++e) sL[(rb + e) * 32 + cg] = acc[e];
  }

  if (l < 16) { // masked softmax, one row per lane (after all WMMAs)
    int pair = pairbase + l;
    int b = pair / MAXR, route = pair % MAXR;
    float rn = state[(size_t)b * SSTRIDE + route];
    float m = -3.4e38f;
    for (int n = 0; n < MAXN; ++n) m = fmaxf(m, sL[l * 32 + n]);
    float sum = 0.0f;
    float p[MAXN];
    for (int n = 0; n < MAXN; ++n) { p[n] = __expf(sL[l * 32 + n] - m); sum += p[n]; }
    float inv = __builtin_amdgcn_rcpf(sum);
    for (int n = 0; n < MAXN; ++n)
      out[(size_t)pair * MAXN + n] = ((float)n < rn) ? p[n] * inv : 0.0f;
  }
}

// ---------------- host: launch pipeline ------------------------------------
extern "C" void kernel_launch(void* const* d_in, const int* in_sizes, int n_in,
                              void* d_out, int out_size, void* d_ws, size_t ws_size,
                              hipStream_t stream) {
  const float* state = (const float*)d_in[0];
  const float* Wc1 = (const float*)d_in[1];
  const float* bc1 = (const float*)d_in[2];
  const float* Wc2 = (const float*)d_in[3];
  const float* bc2 = (const float*)d_in[4];
  const float* Wih0 = (const float*)d_in[5];
  const float* Whh0 = (const float*)d_in[6];
  const float* bih0 = (const float*)d_in[7];
  const float* bhh0 = (const float*)d_in[8];
  const float* Wih1 = (const float*)d_in[9];
  const float* Whh1 = (const float*)d_in[10];
  const float* bih1 = (const float*)d_in[11];
  const float* bhh1 = (const float*)d_in[12];
  const float* Wn1 = (const float*)d_in[13];
  const float* bn1 = (const float*)d_in[14];
  const float* Wn2 = (const float*)d_in[15];
  const float* bn2 = (const float*)d_in[16];
  const float* Wn3 = (const float*)d_in[17];
  const float* bn3 = (const float*)d_in[18];

  char* ws = (char*)d_ws;
  _Float16* wc1s = (_Float16*)(ws + WS_WC1);
  _Float16* wc2s = (_Float16*)(ws + WS_WC2);
  _Float16* wih0s = (_Float16*)(ws + WS_WIH0);
  _Float16* wn1s = (_Float16*)(ws + WS_WN1);
  _Float16* wn2s = (_Float16*)(ws + WS_WN2);
  _Float16* wn3s = (_Float16*)(ws + WS_WN3);
  _Float16* xseq = (_Float16*)(ws + WS_XSEQ);
  float* h2 = (float*)(ws + WS_H2);
  float* meanp = (float*)(ws + WS_MEAN);

  auto sw = [&](const float* W, _Float16* dst, int K, int Kpad, int N, int Npad) {
    int total = Kpad * Npad;
    swizzle_w<<<(total + 255) / 256, 256, 0, stream>>>(W, dst, K, Kpad, N, Npad);
  };
  sw(Wc1, wc1s, 36, 64, 128, 128);
  sw(Wc2, wc2s, 128, 128, 32, 32);
  sw(Wih0, (_Float16*)(ws + WS_WIH0), 32, 32, 384, 384);
  sw(Whh0, (_Float16*)(ws + WS_WHH0), 128, 128, 384, 384);
  sw(Wih1, (_Float16*)(ws + WS_WIH1), 128, 128, 384, 384);
  sw(Whh1, (_Float16*)(ws + WS_WHH1), 128, 128, 384, 384);
  sw(Wn1, wn1s, 256, 256, 256, 256);
  sw(Wn2, wn2s, 256, 256, 128, 128);
  sw(Wn3, wn3s, 128, 128, 24, 32);

  // customer MLP: 294912 rows / (8 waves * 16 rows) = 2304 blocks
  cust_mlp_kernel<<<NROW / 128, 256, (8 * 16 * 64 + 8 * 16 * 128) * 2, stream>>>(
      state, wc1s, bc1, wc2s, bc2, xseq);

  // GRU: 12288 pairs / 64 = 192 blocks x 4 waves; LDS = 312KB weights + 4KB staging
  gru_kernel<<<NPAIR / 64, 128, GRU_WBYTES + 4 * 512 * 2, stream>>>(
      xseq, wih0s, bih0, bhh0, bih1, bhh1, h2);

  mean_kernel<<<Bb, 128, 0, stream>>>(h2, meanp);

  head_kernel<<<NPAIR / 64, 128, 4 * 4096 * 2 * 2 + 4 * 512 * 4, stream>>>(
      h2, meanp, state, wn1s, bn1, wn2s, bn2, wn3s, bn3, (float*)d_out);
}
